// InefficientLinear_38749194944771
// MI455X (gfx1250) — compile-verified, matrix operand order checked by reading, not run on hardware
//
#include <hip/hip_runtime.h>
#include <stdint.h>

typedef __attribute__((ext_vector_type(16))) __bf16 v16bf;
typedef __attribute__((ext_vector_type(8)))  __bf16 v8bf;
typedef __attribute__((ext_vector_type(4)))  __bf16 v4bf;
typedef __attribute__((ext_vector_type(8)))  float  v8f;

constexpr int NROWS = 4096;   // rows of X (M)
constexpr int KDIM  = 2048;   // inner dim
constexpr int NOUT  = 2048;   // rows of W = output cols (N)

constexpr int BM = 128, BN = 128, BK = 32;
constexpr int SLDA = BK + 8;  // 40 bf16 = 80B LDS row stride (bank-conflict padding)

// ---------------- common helpers ----------------

__device__ __forceinline__ v16bf frag_ld(const __bf16* p) {
  v8bf a = *(const v8bf*)(p);
  v8bf b = *(const v8bf*)(p + 16);
  return __builtin_shufflevector(a, b, 0, 1, 2, 3, 4, 5, 6, 7,
                                       8, 9, 10, 11, 12, 13, 14, 15);
}

__device__ __forceinline__ void split4(__bf16* ph, __bf16* pl, float4 v) {
  v4bf h, l;
  float f;
  f = v.x; h[0] = (__bf16)f; l[0] = (__bf16)(f - (float)h[0]);
  f = v.y; h[1] = (__bf16)f; l[1] = (__bf16)(f - (float)h[1]);
  f = v.z; h[2] = (__bf16)f; l[2] = (__bf16)(f - (float)h[2]);
  f = v.w; h[3] = (__bf16)f; l[3] = (__bf16)(f - (float)h[3]);
  *(v4bf*)ph = h;
  *(v4bf*)pl = l;
}

__device__ __forceinline__ v8f wmma_bf16(v16bf a, v16bf b, v8f c) {
  return __builtin_amdgcn_wmma_f32_16x16x32_bf16(
      false, a, false, b, (short)0, c, false, false);
}

// ---------------- pass 1: one-time fp32 -> bf16 hi/lo split ----------------

__global__ __launch_bounds__(256) void split_kernel(const float* __restrict__ S,
                                                    __bf16* __restrict__ H,
                                                    __bf16* __restrict__ L) {
  const int i = blockIdx.x * 256 + threadIdx.x;   // one float4 per thread
  float4 v = ((const float4*)S)[i];
  split4((__bf16*)((v4bf*)H + i), (__bf16*)((v4bf*)L + i), v);
}

// ---------------- pass 2: GEMM on preconverted bf16 planes -----------------
// 128 threads = 4 waves, 2x2 wave grid, 64x64 wave tile, 128x128 block tile.

// CDNA5 async global->LDS copy of one 16B chunk (ASYNCcnt-tracked).
__device__ __forceinline__ void async_cp16(uint32_t lds_byte_off,
                                           const __bf16* g) {
  asm volatile("global_load_async_to_lds_b128 %0, %1, off"
               :: "v"(lds_byte_off), "v"(g) : "memory");
}

__device__ __forceinline__ void wait_async0() {
  asm volatile("s_wait_asynccnt 0x0" ::: "memory");
}

// Issue async fills for one 128x32 bf16 tile of all four planes (128 thd).
__device__ __forceinline__ void async_fill(
    const __bf16* __restrict__ Xh, const __bf16* __restrict__ Xl,
    const __bf16* __restrict__ Wh, const __bf16* __restrict__ Wl,
    int rowBase, int colBase, int k0, int tid,
    __bf16* ah, __bf16* al, __bf16* bh, __bf16* bl) {
#pragma unroll
  for (int t = 0; t < 4; ++t) {
    const int j  = tid + t * 128;        // 512 x 16B chunks per plane
    const int r  = j >> 2;               // row 0..127
    const int c8 = (j & 3) << 3;         // k-offset in bf16 units: 0,8,16,24
    const uint32_t loff = (uint32_t)((r * SLDA + c8) * 2);
    const size_t gx = (size_t)(rowBase + r) * KDIM + k0 + c8;
    const size_t gw = (size_t)(colBase + r) * KDIM + k0 + c8;
    async_cp16((uint32_t)(uintptr_t)ah + loff, Xh + gx);
    async_cp16((uint32_t)(uintptr_t)al + loff, Xl + gx);
    async_cp16((uint32_t)(uintptr_t)bh + loff, Wh + gw);
    async_cp16((uint32_t)(uintptr_t)bl + loff, Wl + gw);
  }
}

// 48 WMMAs on one LDS tile, in 3 spaced passes (each C touched 1x per 16).
__device__ __forceinline__ void compute_tile64(const __bf16* sah, const __bf16* sal,
                                               const __bf16* sbh, const __bf16* sbl,
                                               int wm, int wn, int l15, int kb,
                                               v8f (&C)[4][4]) {
  v16bf Ah[4], Al[4], Bh[4], Bl[4];
#pragma unroll
  for (int at = 0; at < 4; ++at) {
    const int off = (wm * 64 + at * 16 + l15) * SLDA + kb;
    Ah[at] = frag_ld(sah + off);
    Al[at] = frag_ld(sal + off);
  }
#pragma unroll
  for (int bt = 0; bt < 4; ++bt) {
    const int off = (wn * 64 + bt * 16 + l15) * SLDA + kb;
    Bh[bt] = frag_ld(sbh + off);
    Bl[bt] = frag_ld(sbl + off);
  }
  // pass 0: Ah x Bl, pass 1: Al x Bh, pass 2: Ah x Bh
#pragma unroll
  for (int at = 0; at < 4; ++at)
#pragma unroll
    for (int bt = 0; bt < 4; ++bt)
      C[at][bt] = wmma_bf16(Ah[at], Bl[bt], C[at][bt]);
#pragma unroll
  for (int at = 0; at < 4; ++at)
#pragma unroll
    for (int bt = 0; bt < 4; ++bt)
      C[at][bt] = wmma_bf16(Al[at], Bh[bt], C[at][bt]);
#pragma unroll
  for (int at = 0; at < 4; ++at)
#pragma unroll
    for (int bt = 0; bt < 4; ++bt)
      C[at][bt] = wmma_bf16(Ah[at], Bh[bt], C[at][bt]);
}

__global__ __launch_bounds__(128) void gemm_pre_kernel(
    const __bf16* __restrict__ Xh, const __bf16* __restrict__ Xl,
    const __bf16* __restrict__ Wh, const __bf16* __restrict__ Wl,
    const float* __restrict__ B, float* __restrict__ O) {
  __shared__ __bf16 sAh[2][BM * SLDA];
  __shared__ __bf16 sAl[2][BM * SLDA];
  __shared__ __bf16 sBh[2][BN * SLDA];
  __shared__ __bf16 sBl[2][BN * SLDA];

  const int tid  = threadIdx.x;
  const int lane = tid & 31;
  const int wave = tid >> 5;           // 0..3
  const int wm   = wave >> 1;          // 0..1 -> 64-row slice
  const int wn   = wave & 1;           // 0..1 -> 64-col slice
  const int l15  = lane & 15;
  const int kb   = (lane >> 4) << 3;

  const int colBase = blockIdx.x * BN;
  const int rowBase = blockIdx.y * BM;

  v8f C[4][4];
#pragma unroll
  for (int bt = 0; bt < 4; ++bt) {
    float bv = B[colBase + wn * 64 + bt * 16 + l15];
#pragma unroll
    for (int at = 0; at < 4; ++at)
#pragma unroll
      for (int r = 0; r < 8; ++r) C[at][bt][r] = bv;
  }

  // Prologue: async-fill buffer 0 with tile k0 = 0.
  async_fill(Xh, Xl, Wh, Wl, rowBase, colBase, 0, tid,
             sAh[0], sAl[0], sBh[0], sBl[0]);
  wait_async0();
  __syncthreads();

  int p = 0;
  for (int k0 = BK; k0 < KDIM; k0 += BK) {
    const int q = p ^ 1;
    async_fill(Xh, Xl, Wh, Wl, rowBase, colBase, k0, tid,
               sAh[q], sAl[q], sBh[q], sBl[q]);
    compute_tile64(sAh[p], sAl[p], sBh[p], sBl[p], wm, wn, l15, kb, C);
    wait_async0();
    __syncthreads();
    p = q;
  }
  compute_tile64(sAh[p], sAl[p], sBh[p], sBl[p], wm, wn, l15, kb, C);

  // Store: C VGPR r -> M = r (lanes 0-15) or r+8 (lanes 16-31), N = l15.
  const int rsel = (lane >> 4) << 3;
#pragma unroll
  for (int at = 0; at < 4; ++at) {
    const int row0 = rowBase + wm * 64 + at * 16 + rsel;
#pragma unroll
    for (int bt = 0; bt < 4; ++bt) {
      const int col = colBase + wn * 64 + bt * 16 + l15;
#pragma unroll
      for (int r = 0; r < 8; ++r)
        __builtin_nontemporal_store(C[at][bt][r],
                                    O + (size_t)(row0 + r) * NOUT + col);
    }
  }
}

// ---------------- fallback: fused conversion (no workspace needed) ---------
// 256 threads, 64x32 wave tiles, registers-staged fp32 -> bf16 split.

__device__ __forceinline__ void compute_tile32(const __bf16* sah, const __bf16* sal,
                                               const __bf16* sbh, const __bf16* sbl,
                                               int wm, int wn, int l15, int kb,
                                               v8f (&C)[4][2]) {
  v16bf Ah[4], Al[4], Bh[2], Bl[2];
#pragma unroll
  for (int at = 0; at < 4; ++at) {
    const int off = (wm * 64 + at * 16 + l15) * SLDA + kb;
    Ah[at] = frag_ld(sah + off);
    Al[at] = frag_ld(sal + off);
  }
#pragma unroll
  for (int bt = 0; bt < 2; ++bt) {
    const int off = (wn * 32 + bt * 16 + l15) * SLDA + kb;
    Bh[bt] = frag_ld(sbh + off);
    Bl[bt] = frag_ld(sbl + off);
  }
#pragma unroll
  for (int at = 0; at < 4; ++at)
#pragma unroll
    for (int bt = 0; bt < 2; ++bt)
      C[at][bt] = wmma_bf16(Ah[at], Bl[bt], C[at][bt]);
#pragma unroll
  for (int at = 0; at < 4; ++at)
#pragma unroll
    for (int bt = 0; bt < 2; ++bt)
      C[at][bt] = wmma_bf16(Al[at], Bh[bt], C[at][bt]);
#pragma unroll
  for (int at = 0; at < 4; ++at)
#pragma unroll
    for (int bt = 0; bt < 2; ++bt)
      C[at][bt] = wmma_bf16(Ah[at], Bh[bt], C[at][bt]);
}

__device__ __forceinline__ void tile_load(const float* __restrict__ X,
                                          const float* __restrict__ W,
                                          int rowBase, int colBase, int k0,
                                          int tid, float4* rx, float4* rw) {
#pragma unroll
  for (int ii = 0; ii < 4; ++ii) {
    const int i  = tid + ii * 256;
    const int r  = i >> 3;
    const int c4 = (i & 7) << 2;
    rx[ii] = *(const float4*)(X + (size_t)(rowBase + r) * KDIM + k0 + c4);
    rw[ii] = *(const float4*)(W + (size_t)(colBase + r) * KDIM + k0 + c4);
  }
}

__device__ __forceinline__ void tile_store(__bf16* ah, __bf16* al,
                                           __bf16* bh, __bf16* bl,
                                           int tid, const float4* rx,
                                           const float4* rw) {
#pragma unroll
  for (int ii = 0; ii < 4; ++ii) {
    const int i  = tid + ii * 256;
    const int r  = i >> 3;
    const int c4 = (i & 7) << 2;
    split4(&ah[r * SLDA + c4], &al[r * SLDA + c4], rx[ii]);
    split4(&bh[r * SLDA + c4], &bl[r * SLDA + c4], rw[ii]);
  }
}

__global__ __launch_bounds__(256) void linear_fused_kernel(
    const float* __restrict__ X, const float* __restrict__ W,
    const float* __restrict__ B, float* __restrict__ O) {
  __shared__ __bf16 sAh[2][BM * SLDA];
  __shared__ __bf16 sAl[2][BM * SLDA];
  __shared__ __bf16 sBh[2][BN * SLDA];
  __shared__ __bf16 sBl[2][BN * SLDA];

  const int tid  = threadIdx.x;
  const int lane = tid & 31;
  const int wave = tid >> 5;
  const int wm   = wave >> 2;
  const int wn   = wave & 3;
  const int l15  = lane & 15;
  const int kb   = (lane >> 4) << 3;

  const int colBase = blockIdx.x * BN;
  const int rowBase = blockIdx.y * BM;

  v8f C[4][2];
#pragma unroll
  for (int bt = 0; bt < 2; ++bt) {
    float bv = B[colBase + wn * 32 + bt * 16 + l15];
#pragma unroll
    for (int at = 0; at < 4; ++at)
#pragma unroll
      for (int r = 0; r < 8; ++r) C[at][bt][r] = bv;
  }

  float4 rx[4], rw[4];
  tile_load(X, W, rowBase, colBase, 0, tid, rx, rw);
  tile_store(sAh[0], sAl[0], sBh[0], sBl[0], tid, rx, rw);
  __syncthreads();

  int p = 0;
  for (int k0 = BK; k0 < KDIM; k0 += BK) {
    tile_load(X, W, rowBase, colBase, k0, tid, rx, rw);
    compute_tile32(sAh[p], sAl[p], sBh[p], sBl[p], wm, wn, l15, kb, C);
    const int q = p ^ 1;
    tile_store(sAh[q], sAl[q], sBh[q], sBl[q], tid, rx, rw);
    __syncthreads();
    p = q;
  }
  compute_tile32(sAh[p], sAl[p], sBh[p], sBl[p], wm, wn, l15, kb, C);

  const int rsel = (lane >> 4) << 3;
#pragma unroll
  for (int at = 0; at < 4; ++at) {
    const int row0 = rowBase + wm * 64 + at * 16 + rsel;
#pragma unroll
    for (int bt = 0; bt < 2; ++bt) {
      const int col = colBase + wn * 32 + bt * 16 + l15;
#pragma unroll
      for (int r = 0; r < 8; ++r)
        __builtin_nontemporal_store(C[at][bt][r],
                                    O + (size_t)(row0 + r) * NOUT + col);
    }
  }
}

// ---------------- host launch ----------------

extern "C" void kernel_launch(void* const* d_in, const int* in_sizes, int n_in,
                              void* d_out, int out_size, void* d_ws, size_t ws_size,
                              hipStream_t stream) {
  (void)in_sizes; (void)n_in; (void)out_size;
  const float* X = (const float*)d_in[0];
  const float* W = (const float*)d_in[1];
  const float* B = (const float*)d_in[2];
  float* O = (float*)d_out;

  constexpr size_t XN = (size_t)NROWS * KDIM;   // 8388608 elements
  constexpr size_t WN = (size_t)NOUT * KDIM;    // 4194304 elements
  constexpr size_t WS_NEEDED = (XN + WN) * 2 * sizeof(__bf16);  // 48 MiB

  dim3 grid(NOUT / BN, NROWS / BM);  // 16 x 32 blocks

  if (ws_size >= WS_NEEDED) {
    __bf16* Xh = (__bf16*)d_ws;
    __bf16* Xl = Xh + XN;
    __bf16* Wh = Xl + XN;
    __bf16* Wl = Wh + WN;
    split_kernel<<<dim3((unsigned)(XN / 4 / 256)), dim3(256), 0, stream>>>(X, Xh, Xl);
    split_kernel<<<dim3((unsigned)(WN / 4 / 256)), dim3(256), 0, stream>>>(W, Wh, Wl);
    gemm_pre_kernel<<<grid, dim3(128), 0, stream>>>(Xh, Xl, Wh, Wl, B, O);
  } else {
    linear_fused_kernel<<<grid, dim3(256), 0, stream>>>(X, W, B, O);
  }
}
